// TriangleMultiplicationOpti_24292335027054
// MI455X (gfx1250) — compile-verified, matrix operand order checked by reading, not run on hardware
//
#include <hip/hip_runtime.h>
#include <hip/hip_bf16.h>
#include <cstdint>
#include <cstddef>

#define NRES 768
#define CH   128
#define NN   (NRES * NRES)

typedef __attribute__((ext_vector_type(16))) _Float16 v16h;
typedef __attribute__((ext_vector_type(8)))  float    v8f;

// ---- CDNA5 async global->LDS staging via inline asm (ISA 08_async_tensor) ---
// global_load_async_to_lds_b128: VDST = LDS byte offset (low 32 bits of a flat
// shared pointer ARE the LDS offset per ISA 10.2 aperture decode), VADDR = 64b
// global address. Tracked by ASYNCcnt; retires in order -> double buffering
// works with s_wait_asynccnt <= (just-issued count).
__device__ __forceinline__ void stage16(const _Float16* g, _Float16* l) {
  const uint32_t loff = (uint32_t)(size_t)l;
  asm volatile("global_load_async_to_lds_b128 %0, %1, off"
               :
               : "v"(loff), "v"(g)
               : "memory");
}
__device__ __forceinline__ void async_wait0() {
  asm volatile("s_wait_asynccnt 0x0" ::: "memory");
}
__device__ __forceinline__ void async_wait4() {
  asm volatile("s_wait_asynccnt 0x4" ::: "memory");
}

// Load a 16x32 f16 fragment (A layout; identical index map serves B when the
// source is B^T row-major). ISA 05_wmma: lane = (K/8 & 1)*16 + M;
// elements 0..7 -> K = 8h+0..7, elements 8..15 -> K = 16+8h+0..7.
__device__ __forceinline__ v16h load_frag16(const _Float16* __restrict__ base,
                                            int stride) {
  const int lane = threadIdx.x & 31;
  const int m = lane & 15;
  const int h = lane >> 4;
  const _Float16* r = base + m * stride + 8 * h;
  v16h f;
#pragma unroll
  for (int e = 0; e < 8; ++e) f[e] = r[e];
#pragma unroll
  for (int e = 0; e < 8; ++e) f[8 + e] = r[16 + e];
  return f;
}

__device__ __forceinline__ v8f wmma_f16(v16h a, v16h b, v8f c) {
  return __builtin_amdgcn_wmma_f32_16x16x32_f16(false, a, false, b, (short)0, c,
                                                false, false);
}

__device__ __forceinline__ float sigmoidf_(float x) {
  return 1.0f / (1.0f + __expf(-x));
}

// ---------------- kernel W: fp32 weights -> f16, [6][128][128] ----------------
__global__ __launch_bounds__(256) void k_cvt_weights(
    const float* __restrict__ w0, const float* __restrict__ w1,
    const float* __restrict__ w2, const float* __restrict__ w3,
    const float* __restrict__ w4, const float* __restrict__ w5,
    _Float16* __restrict__ Wh) {
  const int i = blockIdx.x * 256 + threadIdx.x;
  if (i >= 6 * CH * CH) return;
  const float* srcs[6] = {w0, w1, w2, w3, w4, w5};
  Wh[i] = (_Float16)srcs[i >> 14][i & (CH * CH - 1)];
}

// ---- kernel A: input LN + 5 projections via WMMA + mask/sigmoid fusion ------
// Block: 256 threads (8 waves), 128 flattened (i,k) rows per block.
// Outputs: Lt/Rt f16 in [c][i*N+k] layout, gate f16 in [p][c] layout.
__global__ __launch_bounds__(256) void k_ln_proj_gate(
    const float* __restrict__ act, const float* __restrict__ mask,
    const float* __restrict__ lnw, const float* __restrict__ lnb,
    const _Float16* __restrict__ Wh, const float* __restrict__ blp,
    const float* __restrict__ brp, const float* __restrict__ blg,
    const float* __restrict__ brg, const float* __restrict__ bga,
    _Float16* __restrict__ Lt, _Float16* __restrict__ Rt,
    _Float16* __restrict__ Gt) {
  __shared__ _Float16 xh[128 * CH];  // 32 KB: LN'd activations, f16
  __shared__ float mk[128];
  const int t = threadIdx.x;
  const size_t p0 = (size_t)blockIdx.x * 128;

  {  // LayerNorm: 2 threads/row, 64 channels each, pair-combine via shfl_xor.
    const int row = t >> 1, half = t & 1;
    const float* xr = act + (p0 + row) * CH + 64 * half;
    float s1 = 0.f, s2 = 0.f;
#pragma unroll
    for (int c = 0; c < 64; c += 4) {
      float4 v = *(const float4*)(xr + c);
      s1 += v.x + v.y + v.z + v.w;
      s2 += v.x * v.x + v.y * v.y + v.z * v.z + v.w * v.w;
    }
    s1 += __shfl_xor(s1, 1, 32);
    s2 += __shfl_xor(s2, 1, 32);
    const float mu = s1 * (1.f / CH);
    const float rs = rsqrtf(s2 * (1.f / CH) - mu * mu + 1e-5f);
#pragma unroll
    for (int c = 0; c < 64; c += 4) {
      float4 v = *(const float4*)(xr + c);
      const int cc = 64 * half + c;
      xh[row * CH + cc + 0] = (_Float16)((v.x - mu) * rs * lnw[cc + 0] + lnb[cc + 0]);
      xh[row * CH + cc + 1] = (_Float16)((v.y - mu) * rs * lnw[cc + 1] + lnb[cc + 1]);
      xh[row * CH + cc + 2] = (_Float16)((v.z - mu) * rs * lnw[cc + 2] + lnb[cc + 2]);
      xh[row * CH + cc + 3] = (_Float16)((v.w - mu) * rs * lnw[cc + 3] + lnb[cc + 3]);
    }
    if (t < 128) mk[t] = mask[p0 + t];
  }
  __syncthreads();

  const int lane = t & 31, wv = t >> 5;
  const int nl = lane & 15, h = lane >> 4;
  v16h a[4];
#pragma unroll
  for (int kk = 0; kk < 4; ++kk)
    a[kk] = load_frag16(xh + (16 * wv) * CH + 32 * kk, CH);
  float mrow[8];
#pragma unroll
  for (int r = 0; r < 8; ++r) mrow[r] = mk[16 * wv + 8 * h + r];
  const size_t prow0 = p0 + (size_t)(16 * wv + 8 * h);

  for (int nt = 0; nt < 8; ++nt) {
    const int cch = nt * 16 + nl;
    const _Float16* wbase = Wh + (size_t)(nt * 16) * CH;
    {  // left = mask * (x@Wlp^T + b) * sigmoid(x@Wlg^T + b)
      v8f ap = {}, ag = {};
#pragma unroll
      for (int kk = 0; kk < 4; ++kk) {
        ap = wmma_f16(a[kk], load_frag16(wbase + 0 * CH * CH + 32 * kk, CH), ap);
        ag = wmma_f16(a[kk], load_frag16(wbase + 2 * CH * CH + 32 * kk, CH), ag);
      }
      const float bp = blp[cch], bg = blg[cch];
#pragma unroll
      for (int r = 0; r < 8; ++r) {
        const float v = mrow[r] * (ap[r] + bp) * sigmoidf_(ag[r] + bg);
        Lt[(size_t)cch * NN + prow0 + r] = (_Float16)v;
      }
    }
    {  // right
      v8f ap = {}, ag = {};
#pragma unroll
      for (int kk = 0; kk < 4; ++kk) {
        ap = wmma_f16(a[kk], load_frag16(wbase + 1 * CH * CH + 32 * kk, CH), ap);
        ag = wmma_f16(a[kk], load_frag16(wbase + 3 * CH * CH + 32 * kk, CH), ag);
      }
      const float bp = brp[cch], bg = brg[cch];
#pragma unroll
      for (int r = 0; r < 8; ++r) {
        const float v = mrow[r] * (ap[r] + bp) * sigmoidf_(ag[r] + bg);
        Rt[(size_t)cch * NN + prow0 + r] = (_Float16)v;
      }
    }
    {  // output gate = sigmoid(x@Wg^T + b), stored [p][c]
      v8f ag = {};
#pragma unroll
      for (int kk = 0; kk < 4; ++kk)
        ag = wmma_f16(a[kk], load_frag16(wbase + 4 * CH * CH + 32 * kk, CH), ag);
      const float bg = bga[cch];
#pragma unroll
      for (int r = 0; r < 8; ++r)
        Gt[(prow0 + r) * CH + cch] = (_Float16)sigmoidf_(ag[r] + bg);
    }
  }
}

// ---- kernel B: per-channel GEMM  OUT_c = L_c * R_c^T --------------------
// 128x128 block tile, K-step 32, double-buffered async LDS staging.
// Wave w: rows [32*(w&3), +32) x cols [64*(w>>2), +64) -> 2x4 accumulators,
// 8 WMMAs per K-step from 2 A-frags + 4 B-frags.
__global__ __launch_bounds__(256) void k_triangle(
    const _Float16* __restrict__ Lt, const _Float16* __restrict__ Rt,
    float* __restrict__ tmp) {
  __shared__ _Float16 Ls[2][128 * 32];  // 2 x 8 KB
  __shared__ _Float16 Rs[2][128 * 32];  // 2 x 8 KB
  const int t = threadIdx.x;
  const int c = blockIdx.y;
  const int i0 = (blockIdx.x / 6) * 128;
  const int j0 = (blockIdx.x % 6) * 128;
  const size_t cbase = (size_t)c * NN;
  const int wv = t >> 5, lane = t & 31, h = lane >> 4, nl = lane & 15;
  const int rw = wv & 3;   // row-pair tile: rows 32*rw .. 32*rw+31
  const int cw = wv >> 2;  // col-quad tile: cols 64*cw .. 64*cw+63

  // Staging map: 128 rows x 32 halfs; 16B chunks; 2 chunks/tensor/thread.
  const int srow = t >> 2;
  const int scol = (t & 3) * 8;
  const _Float16* glb = Lt + cbase + (size_t)i0 * NRES;
  const _Float16* grb = Rt + cbase + (size_t)j0 * NRES;

#define STAGE(buf, k0)                                                        \
  do {                                                                        \
    stage16(glb + (size_t)srow * NRES + (k0) + scol,                          \
            &Ls[buf][srow * 32 + scol]);                                      \
    stage16(glb + (size_t)(srow + 64) * NRES + (k0) + scol,                   \
            &Ls[buf][(srow + 64) * 32 + scol]);                               \
    stage16(grb + (size_t)srow * NRES + (k0) + scol,                          \
            &Rs[buf][srow * 32 + scol]);                                      \
    stage16(grb + (size_t)(srow + 64) * NRES + (k0) + scol,                   \
            &Rs[buf][(srow + 64) * 32 + scol]);                               \
  } while (0)

  v8f acc[2][4] = {};
  int buf = 0;
  STAGE(0, 0);
  for (int k0 = 0; k0 < NRES; k0 += 32) {
    if (k0 + 32 < NRES) {
      STAGE(buf ^ 1, k0 + 32);  // prefetch next K-step (4 async ops)
      async_wait4();            // retire current K-step's 4 (in-order)
    } else {
      async_wait0();
    }
    __syncthreads();
    const _Float16* lb = Ls[buf];
    const _Float16* rb = Rs[buf];
    const v16h a0 = load_frag16(lb + (32 * rw) * 32, 32);
    const v16h a1 = load_frag16(lb + (32 * rw + 16) * 32, 32);
#pragma unroll
    for (int nj = 0; nj < 4; ++nj) {
      const v16h b = load_frag16(rb + (16 * (4 * cw + nj)) * 32, 32);
      acc[0][nj] = wmma_f16(a0, b, acc[0][nj]);
      acc[1][nj] = wmma_f16(a1, b, acc[1][nj]);
    }
    __syncthreads();  // all waves done reading buf before it is restaged
    buf ^= 1;
  }
#undef STAGE

  const int irow = i0 + 32 * rw + 8 * h;
#pragma unroll
  for (int si = 0; si < 2; ++si) {
#pragma unroll
    for (int nj = 0; nj < 4; ++nj) {
      const int jj = j0 + 16 * (4 * cw + nj) + nl;
#pragma unroll
      for (int r = 0; r < 8; ++r) {
        tmp[((size_t)(irow + 16 * si + r) * NRES + jj) * CH + c] =
            acc[si][nj][r];
      }
    }
  }
}

// ---- kernel C: center LN + out_proj (WMMA) + sigmoid-gate multiply ----------
__global__ __launch_bounds__(256) void k_epilogue(
    const float* __restrict__ tmp, const float* __restrict__ cw,
    const float* __restrict__ cb, const _Float16* __restrict__ Wh,
    const float* __restrict__ ob, const _Float16* __restrict__ Gt,
    float* __restrict__ out) {
  __shared__ _Float16 xh[128 * CH];
  const int t = threadIdx.x;
  const size_t p0 = (size_t)blockIdx.x * 128;
  {
    const int row = t >> 1, half = t & 1;
    const float* xr = tmp + (p0 + row) * CH + 64 * half;
    float s1 = 0.f, s2 = 0.f;
#pragma unroll
    for (int c = 0; c < 64; c += 4) {
      float4 v = *(const float4*)(xr + c);
      s1 += v.x + v.y + v.z + v.w;
      s2 += v.x * v.x + v.y * v.y + v.z * v.z + v.w * v.w;
    }
    s1 += __shfl_xor(s1, 1, 32);
    s2 += __shfl_xor(s2, 1, 32);
    const float mu = s1 * (1.f / CH);
    const float rs = rsqrtf(s2 * (1.f / CH) - mu * mu + 1e-5f);
#pragma unroll
    for (int c = 0; c < 64; c += 4) {
      float4 v = *(const float4*)(xr + c);
      const int cc = 64 * half + c;
      xh[row * CH + cc + 0] = (_Float16)((v.x - mu) * rs * cw[cc + 0] + cb[cc + 0]);
      xh[row * CH + cc + 1] = (_Float16)((v.y - mu) * rs * cw[cc + 1] + cb[cc + 1]);
      xh[row * CH + cc + 2] = (_Float16)((v.z - mu) * rs * cw[cc + 2] + cb[cc + 2]);
      xh[row * CH + cc + 3] = (_Float16)((v.w - mu) * rs * cw[cc + 3] + cb[cc + 3]);
    }
  }
  __syncthreads();
  const int lane = t & 31, wv = t >> 5, nl = lane & 15, h = lane >> 4;
  v16h a[4];
#pragma unroll
  for (int kk = 0; kk < 4; ++kk)
    a[kk] = load_frag16(xh + (16 * wv) * CH + 32 * kk, CH);
  const size_t prow0 = p0 + (size_t)(16 * wv + 8 * h);
  const _Float16* Wop = Wh + (size_t)5 * CH * CH;
  for (int nt = 0; nt < 8; ++nt) {
    const int cch = nt * 16 + nl;
    v8f acc = {};
#pragma unroll
    for (int kk = 0; kk < 4; ++kk)
      acc = wmma_f16(a[kk], load_frag16(Wop + (nt * 16) * CH + 32 * kk, CH), acc);
    const float b = ob[cch];
#pragma unroll
    for (int r = 0; r < 8; ++r) {
      const size_t p = prow0 + r;
      out[p * CH + cch] = (acc[r] + b) * (float)Gt[p * CH + cch];
    }
  }
}

// -----------------------------------------------------------------------------
extern "C" void kernel_launch(void* const* d_in, const int* in_sizes, int n_in,
                              void* d_out, int out_size, void* d_ws,
                              size_t ws_size, hipStream_t stream) {
  const float* act = (const float*)d_in[0];
  const float* mask = (const float*)d_in[1];
  const float* lnw = (const float*)d_in[2];
  const float* lnb = (const float*)d_in[3];
  const float* lpw = (const float*)d_in[4];
  const float* lpb = (const float*)d_in[5];
  const float* rpw = (const float*)d_in[6];
  const float* rpb = (const float*)d_in[7];
  const float* lgw = (const float*)d_in[8];
  const float* lgb = (const float*)d_in[9];
  const float* rgw = (const float*)d_in[10];
  const float* rgb = (const float*)d_in[11];
  const float* clw = (const float*)d_in[12];
  const float* clb = (const float*)d_in[13];
  const float* opw = (const float*)d_in[14];
  const float* opb = (const float*)d_in[15];
  const float* gaw = (const float*)d_in[16];
  const float* gab = (const float*)d_in[17];
  float* out = (float*)d_out;

  // Workspace layout (~755 MB):
  //   Wh  f16 [6][128][128]        196608 B
  //   Lt  f16 [C][N*N]          150994944 B
  //   Rt  f16 [C][N*N]          150994944 B
  //   Gt  f16 [N*N][C]          150994944 B
  //   tmp f32 [N*N][C]          301989888 B
  char* ws = (char*)d_ws;
  _Float16* Wh = (_Float16*)ws;
  size_t off = 196608;
  _Float16* Lt = (_Float16*)(ws + off);
  off += (size_t)NN * CH * 2;
  _Float16* Rt = (_Float16*)(ws + off);
  off += (size_t)NN * CH * 2;
  _Float16* Gt = (_Float16*)(ws + off);
  off += (size_t)NN * CH * 2;
  float* tmp = (float*)(ws + off);

  k_cvt_weights<<<(6 * CH * CH + 255) / 256, 256, 0, stream>>>(
      lpw, rpw, lgw, rgw, gaw, opw, Wh);
  k_ln_proj_gate<<<NN / 128, 256, 0, stream>>>(act, mask, lnw, lnb, Wh, lpb,
                                               rpb, lgb, rgb, gab, Lt, Rt, Gt);
  k_triangle<<<dim3(36, 128), 256, 0, stream>>>(Lt, Rt, tmp);
  k_epilogue<<<NN / 128, 256, 0, stream>>>(tmp, clw, clb, Wh, opb, Gt, out);
}
